// MultiHeadAttention_25735444038457
// MI455X (gfx1250) — matrix operand order repaired and checked by reading, not compiled, not run on hardware
//
#include <hip/hip_runtime.h>

// ---------------------------------------------------------------------------
// MultiHeadAttention for MI455X (gfx1250): f16 WMMA everywhere, flash-fused
// attention, async global->LDS double-buffered staging.
//   B=2, N=2048, E=1024, H=16, D=64
// Pipeline:
//   0) x,y -> f16; Wq/Wk/Wv/Wo -> f16 transposed [n][k]   (prep kernels)
//   1) Q=x@Wq+bq, K=y@Wk+bk (f16 [b,h,n,d]); V=y@Wv+bv (f16 [b,h,d,n])
//   2) flash attention: softmax(QK^T)/sqrt(E) @ V  -> o16 [b,n,e] f16
//   3) out = o16@Wo + bo (f32)
// ---------------------------------------------------------------------------

typedef __attribute__((ext_vector_type(16))) _Float16 v16h;
typedef __attribute__((ext_vector_type(8)))  float    v8f;
typedef __attribute__((ext_vector_type(4)))  int      v4i;
typedef _Float16 half_t;

constexpr int kB = 2;
constexpr int kN = 2048;
constexpr int kE = 1024;
constexpr int kH = 16;
constexpr int kD = 64;
constexpr int kM = kB * kN;          // 4096 rows in the GEMMs
constexpr float kSqrtE = 32.0f;      // sqrt(1024); applied AFTER softmax (quirk)

#if defined(__gfx1250__) && __has_builtin(__builtin_amdgcn_global_load_async_to_lds_b128)
#define USE_ASYNC_LDS 1
#else
#define USE_ASYNC_LDS 0
#endif

#if defined(__gfx1250__) && __has_builtin(__builtin_amdgcn_update_dpp)
#define USE_DPP 1
#else
#define USE_DPP 0
#endif

// A/B fragment: 16 halves per lane = two 16-byte segments.
union Frag {
  v16h v;
  uint4 q[2];
  half_t h[16];
};

__device__ __forceinline__ v8f wmma16(const Frag& a, const Frag& b, v8f c) {
  return __builtin_amdgcn_wmma_f32_16x16x32_f16(false, a.v, false, b.v,
                                                (short)0, c, false, false);
}

// 16-byte global -> LDS copy; async (ASYNCcnt) when the builtin exists.
__device__ __forceinline__ void copy16_to_lds(half_t* ldst, const half_t* gsrc) {
#if USE_ASYNC_LDS
  typedef __attribute__((address_space(1))) v4i as1_v4i;
  typedef __attribute__((address_space(3))) v4i as3_v4i;
  __builtin_amdgcn_global_load_async_to_lds_b128(
      (as1_v4i*)(half_t*)gsrc,   // const_cast, then addrspacecast to global
      (as3_v4i*)ldst,            // addrspacecast generic -> LDS
      0, 0);
#else
  *(uint4*)ldst = *(const uint4*)gsrc;
#endif
}

__device__ __forceinline__ void wait_async_copies() {
#if USE_ASYNC_LDS
#if __has_builtin(__builtin_amdgcn_s_wait_asynccnt)
  __builtin_amdgcn_s_wait_asynccnt(0);
#else
  asm volatile("s_wait_asynccnt 0x0" ::: "memory");
#endif
#endif
}

// Max across the 16 lanes of a half-wave (VALU DPP; falls back to shuffles).
__device__ __forceinline__ float max16(float x) {
#if USE_DPP
  int v, y;
  v = __builtin_bit_cast(int, x);
  y = __builtin_amdgcn_update_dpp(0, v, 0xB1, 0xF, 0xF, true);   // quad xor1
  x = fmaxf(x, __builtin_bit_cast(float, y));
  v = __builtin_bit_cast(int, x);
  y = __builtin_amdgcn_update_dpp(0, v, 0x4E, 0xF, 0xF, true);   // quad xor2
  x = fmaxf(x, __builtin_bit_cast(float, y));
  v = __builtin_bit_cast(int, x);
  y = __builtin_amdgcn_update_dpp(0, v, 0x141, 0xF, 0xF, true);  // half mirror
  x = fmaxf(x, __builtin_bit_cast(float, y));
  v = __builtin_bit_cast(int, x);
  y = __builtin_amdgcn_update_dpp(0, v, 0x140, 0xF, 0xF, true);  // row mirror
  x = fmaxf(x, __builtin_bit_cast(float, y));
  return x;
#else
  for (int s = 1; s < 16; s <<= 1) x = fmaxf(x, __shfl_xor(x, s, 32));
  return x;
#endif
}

// ---------------------------------------------------------------------------
// Prep: f32 -> f16 (4 elems/thread).
// ---------------------------------------------------------------------------
__global__ __launch_bounds__(256) void cvt_f32_f16(const float* __restrict__ in,
                                                   half_t* __restrict__ out) {
  const size_t i = ((size_t)blockIdx.x * 256 + threadIdx.x) * 4;
  float4 f = *(const float4*)(in + i);
  alignas(8) half_t h4[4] = {(half_t)f.x, (half_t)f.y, (half_t)f.z, (half_t)f.w};
  *(uint2*)(out + i) = *(const uint2*)h4;
}

// Prep: W[k][n] f32 -> Wt[n][k] f16, 32x32 LDS tile transpose.
__global__ __launch_bounds__(256) void transpose_cvt_w(const float* __restrict__ W,
                                                       half_t* __restrict__ Wt) {
  __shared__ float sT[32][33];
  const int t  = threadIdx.x;
  const int k0 = blockIdx.y * 32, n0 = blockIdx.x * 32;
  const int r  = t >> 3, c4 = (t & 7) * 4;
  float4 f = *(const float4*)(W + (size_t)(k0 + r) * kE + n0 + c4);
  sT[r][c4 + 0] = f.x; sT[r][c4 + 1] = f.y;
  sT[r][c4 + 2] = f.z; sT[r][c4 + 3] = f.w;
  __syncthreads();
  alignas(8) half_t h4[4];
#pragma unroll
  for (int i = 0; i < 4; ++i) h4[i] = (half_t)sT[c4 + i][r];
  *(uint2*)(Wt + (size_t)(n0 + r) * kE + k0 + c4) = *(const uint2*)h4;
}

// ---------------------------------------------------------------------------
// GEMM: Out[M,E] = A[M,K] x Wt[N,K]^T + bias. All-f16 operands, async-staged,
// double-buffered. Block tile 64x128, 8 waves (4x2) of 16x64. K-step 32.
// OUT_MODE: 0 = f32 [M,E];  1 = f16 [b,h,n,d];  2 = f16 [b,h,d,n] (V).
// ---------------------------------------------------------------------------
template <int OUT_MODE>
__global__ __launch_bounds__(256) void gemm_bias_wmma(
    const half_t* __restrict__ A, const half_t* __restrict__ Wt,
    const float* __restrict__ bias, void* __restrict__ Out) {
  __shared__ half_t sA[2][64][40];    // [m][k], 80B rows (16B aligned)
  __shared__ half_t sB[2][128][40];   // Wt tile: [n][k]

  const int t     = threadIdx.x;
  const int wave  = t >> 5;
  const int lane  = t & 31;
  const int lmod  = lane & 15;
  const int lhalf = lane >> 4;
  const int wrow  = wave & 3;
  const int wcol  = wave >> 2;
  const int m0 = blockIdx.y * 64;
  const int n0 = blockIdx.x * 128;

  v8f acc[4] = {};

  auto stage = [&](int buf, int k0) {
    // A tile 64x32 halves: one 16B chunk/thread.
    const int ai = t >> 2, aj = (t & 3) * 8;
    copy16_to_lds(&sA[buf][ai][aj], A + (size_t)(m0 + ai) * kE + k0 + aj);
    // B tile 128x32 halves: two 16B chunks/thread.
#pragma unroll
    for (int c = 0; c < 2; ++c) {
      const int f  = t * 2 + c;        // 0..511
      const int bn = f >> 2, bo = (f & 3) * 8;
      copy16_to_lds(&sB[buf][bn][bo], Wt + (size_t)(n0 + bn) * kE + k0 + bo);
    }
  };

  stage(0, 0);
  constexpr int KT = kE / 32;
  for (int it = 0; it < KT; ++it) {
    const int cur = it & 1;
    wait_async_copies();
    __syncthreads();
    if (it + 1 < KT) stage(cur ^ 1, (it + 1) * 32);

    Frag a;
    {
      const half_t* ar = &sA[cur][wrow * 16 + lmod][0];
      a.q[0] = *(const uint4*)(ar + lhalf * 8);
      a.q[1] = *(const uint4*)(ar + 16 + lhalf * 8);
    }
#pragma unroll
    for (int nt = 0; nt < 4; ++nt) {
      Frag bf;
      const half_t* br = &sB[cur][wcol * 64 + nt * 16 + lmod][lhalf * 8];
      bf.q[0] = *(const uint4*)br;
      bf.q[1] = *(const uint4*)(br + 16);
      acc[nt] = wmma16(a, bf, acc[nt]);
    }
  }

  // Epilogue. C layout: lane holds col n=L%16 of rows M=(L/16)*8+r.
#pragma unroll
  for (int nt = 0; nt < 4; ++nt) {
    const int   e  = n0 + wcol * 64 + nt * 16 + lmod;
    const float bv = bias[e];
#pragma unroll
    for (int r = 0; r < 8; ++r) {
      const int   gm  = m0 + wrow * 16 + lhalf * 8 + r;
      const float val = acc[nt][r] + bv;
      if constexpr (OUT_MODE == 0) {
        ((float*)Out)[(size_t)gm * kE + e] = val;
      } else {
        const int bb = gm / kN, ns = gm % kN;
        const int hh = e >> 6, dd = e & 63;
        if constexpr (OUT_MODE == 1) {
          ((half_t*)Out)[(((size_t)bb * kH + hh) * kN + ns) * kD + dd] =
              (half_t)val;
        } else {  // transposed: [b,h,d,n]
          ((half_t*)Out)[(((size_t)bb * kH + hh) * kD + dd) * kN + ns] =
              (half_t)val;
        }
      }
    }
  }
}

// ---------------------------------------------------------------------------
// Flash attention. grid=(N/128, B*H); 8 waves x 16 query rows.
// K tiles [32 keys][64 d] and V^T tiles [64 d][32 keys] are straight copies
// from global (V was stored pre-transposed) -> async double-buffered staging.
// Row max: DPP reduce. Row sum: P @ ones via one extra WMMA.
// ---------------------------------------------------------------------------
__global__ __launch_bounds__(256) void flash_attn_wmma(
    const half_t* __restrict__ q16, const half_t* __restrict__ k16,
    const half_t* __restrict__ vt16, half_t* __restrict__ o16 /*[B,N,E]*/) {
  __shared__ half_t sK[2][32][72];    // [key][d], 144B rows (conflict-free)
  __shared__ half_t sVt[2][64][40];   // [d][key], 80B rows
  __shared__ half_t sP[8][16][40];    // per-wave P tile [q][key]

  const int t     = threadIdx.x;
  const int wave  = t >> 5;
  const int lane  = t & 31;
  const int lmod  = lane & 15;
  const int lhalf = lane >> 4;

  const int bh = blockIdx.y;
  const int b  = bh >> 4;
  const int h  = bh & 15;
  const int q0 = blockIdx.x * 128 + wave * 16;

  const half_t* qp  = q16  + (size_t)bh * kN * kD;   // [n][d]
  const half_t* kp  = k16  + (size_t)bh * kN * kD;   // [n][d]
  const half_t* vtp = vt16 + (size_t)bh * kD * kN;   // [d][n]

  // Q fragments (2 d-chunks of 32) for this wave's 16 rows, in registers.
  Frag qa[2];
  {
    const half_t* qr = qp + (size_t)(q0 + lmod) * kD;
    const int kh = lhalf * 8;
    qa[0].q[0] = *(const uint4*)(qr + kh);
    qa[0].q[1] = *(const uint4*)(qr + 16 + kh);
    qa[1].q[0] = *(const uint4*)(qr + 32 + kh);
    qa[1].q[1] = *(const uint4*)(qr + 48 + kh);
  }

  Frag ones;
#pragma unroll
  for (int i = 0; i < 16; ++i) ones.h[i] = (half_t)1.0f;

  v8f oacc[4] = {};
  float mrun[8], lrun[8];
#pragma unroll
  for (int r = 0; r < 8; ++r) { mrun[r] = -3.0e38f; lrun[r] = 0.0f; }

  auto stage = [&](int buf, int kt) {
    const int krow = t >> 3, koff = (t & 7) * 8;   // K: 32x64 halves
    copy16_to_lds(&sK[buf][krow][koff],
                  kp + (size_t)(kt + krow) * kD + koff);
    const int vrow = t >> 2, voff = (t & 3) * 8;   // Vt: 64x32 halves
    copy16_to_lds(&sVt[buf][vrow][voff],
                  vtp + (size_t)vrow * kN + kt + voff);
  };

  stage(0, 0);
  constexpr int TILES = kN / 32;
  for (int it = 0; it < TILES; ++it) {
    const int cur = it & 1;
    wait_async_copies();
    __syncthreads();
    if (it + 1 < TILES) stage(cur ^ 1, (it + 1) * 32);

    // ---- energy: e0 = q @ K[0:16]^T, e1 = q @ K[16:32]^T ----
    v8f e0 = {}, e1 = {};
#pragma unroll
    for (int dc = 0; dc < 2; ++dc) {
      Frag b0, b1;   // B[d][key]: lane col = key, halves = d in table order
      const half_t* p0 = &sK[cur][lmod][dc * 32 + lhalf * 8];
      b0.q[0] = *(const uint4*)p0;
      b0.q[1] = *(const uint4*)(p0 + 16);
      const half_t* p1 = &sK[cur][16 + lmod][dc * 32 + lhalf * 8];
      b1.q[0] = *(const uint4*)p1;
      b1.q[1] = *(const uint4*)(p1 + 16);
      e0 = wmma16(qa[dc], b0, e0);
      e1 = wmma16(qa[dc], b1, e1);
    }

    // ---- online softmax: DPP row-max, exp, rescale ----
    float corr[8];
#pragma unroll
    for (int r = 0; r < 8; ++r) {
      const float mx   = max16(fmaxf(e0[r], e1[r]));
      const float mnew = fmaxf(mrun[r], mx);
      corr[r] = __expf(mrun[r] - mnew);
      e0[r]   = __expf(e0[r] - mnew);
      e1[r]   = __expf(e1[r] - mnew);
      mrun[r] = mnew;
    }
#pragma unroll
    for (int c = 0; c < 4; ++c)
#pragma unroll
      for (int r = 0; r < 8; ++r) oacc[c][r] *= corr[r];

    // ---- C-layout -> A-fragment layout for P via per-wave LDS tile ----
#pragma unroll
    for (int r = 0; r < 8; ++r) {
      const int row = lhalf * 8 + r;
      sP[wave][row][lmod]      = (half_t)e0[r];
      sP[wave][row][16 + lmod] = (half_t)e1[r];
    }
    Frag pa;   // LDS ops are in-order within a wave
    {
      const half_t* pr = &sP[wave][lmod][lhalf * 8];
      pa.q[0] = *(const uint4*)pr;
      pa.q[1] = *(const uint4*)(pr + 16);
    }

    // ---- row sums via WMMA: P @ ones -> sum replicated across columns ----
    {
      v8f z = {};
      v8f rs = wmma16(pa, ones, z);
#pragma unroll
      for (int r = 0; r < 8; ++r) lrun[r] = lrun[r] * corr[r] + rs[r];
    }

    // ---- O += P @ V (4 d-chunks of 16) ----
#pragma unroll
    for (int c = 0; c < 4; ++c) {
      Frag vb;  // B[key][d]: lane col = d, halves = keys in table order
      const half_t* vr = &sVt[cur][c * 16 + lmod][lhalf * 8];
      vb.q[0] = *(const uint4*)vr;
      vb.q[1] = *(const uint4*)(vr + 16);
      oacc[c] = wmma16(pa, vb, oacc[c]);
    }
    // single barrier/iteration: next iteration's top barrier guards reuse
  }

  // ---- epilogue: /= (rowsum * sqrt(E)); store [b, n, h*D+d] f16 ----
#pragma unroll
  for (int c = 0; c < 4; ++c) {
    const int d = c * 16 + lmod;
#pragma unroll
    for (int r = 0; r < 8; ++r) {
      const int   qrow = q0 + lhalf * 8 + r;
      const float val  = oacc[c][r] / (lrun[r] * kSqrtE);
      o16[((size_t)b * kN + qrow) * kE + h * kD + d] = (half_t)val;
    }
  }
}

// ---------------------------------------------------------------------------
extern "C" void kernel_launch(void* const* d_in, const int* in_sizes, int n_in,
                              void* d_out, int out_size, void* d_ws,
                              size_t ws_size, hipStream_t stream) {
  (void)in_sizes; (void)n_in; (void)out_size; (void)ws_size;
  const float* x  = (const float*)d_in[0];
  const float* y  = (const float*)d_in[1];
  const float* Wq = (const float*)d_in[2];
  const float* bq = (const float*)d_in[3];
  const float* Wk = (const float*)d_in[4];
  const float* bk = (const float*)d_in[5];
  const float* Wv = (const float*)d_in[6];
  const float* bv = (const float*)d_in[7];
  const float* Wo = (const float*)d_in[8];
  const float* bo = (const float*)d_in[9];

  // Workspace (f16): x16,y16 [M,E]; Wt x4 [E,E]; q,k [b,h,n,d]; v [b,h,d,n];
  // o16 [B,N,E].  Total 28M halves = 56 MB.
  const size_t mE   = (size_t)kM * kE;          // 4 Mi
  const size_t eE   = (size_t)kE * kE;          // 1 Mi
  half_t* x16 = (half_t*)d_ws;
  half_t* y16 = x16 + mE;
  half_t* Wqt = y16 + mE;
  half_t* Wkt = Wqt + eE;
  half_t* Wvt = Wkt + eE;
  half_t* Wot = Wvt + eE;
  half_t* q16 = Wot + eE;
  half_t* k16 = q16 + mE;
  half_t* v16 = k16 + mE;   // stored transposed [b,h,d,n]
  half_t* o16 = v16 + mE;

  dim3 blk(256);
  // prep
  cvt_f32_f16<<<dim3((unsigned)(mE / 1024)), blk, 0, stream>>>(x, x16);
  cvt_f32_f16<<<dim3((unsigned)(mE / 1024)), blk, 0, stream>>>(y, y16);
  dim3 gT(kE / 32, kE / 32);
  transpose_cvt_w<<<gT, blk, 0, stream>>>(Wq, Wqt);
  transpose_cvt_w<<<gT, blk, 0, stream>>>(Wk, Wkt);
  transpose_cvt_w<<<gT, blk, 0, stream>>>(Wv, Wvt);
  transpose_cvt_w<<<gT, blk, 0, stream>>>(Wo, Wot);

  // projections
  dim3 gProj(kE / 128, kM / 64);   // (8, 64)
  gemm_bias_wmma<1><<<gProj, blk, 0, stream>>>(x16, Wqt, bq, q16);
  gemm_bias_wmma<1><<<gProj, blk, 0, stream>>>(y16, Wkt, bk, k16);
  gemm_bias_wmma<2><<<gProj, blk, 0, stream>>>(y16, Wvt, bv, v16);

  // attention
  dim3 gAtt(kN / 128, kB * kH);    // (16, 32)
  flash_attn_wmma<<<gAtt, blk, 0, stream>>>(q16, k16, v16, o16);

  // output projection
  gemm_bias_wmma<0><<<gProj, blk, 0, stream>>>(o16, Wot, bo, d_out);
}
